// Attention_88510686036327
// MI455X (gfx1250) — matrix operand order repaired and checked by reading, not compiled, not run on hardware
//
#include <hip/hip_runtime.h>
#include <hip/hip_bf16.h>

// ---------------------------------------------------------------------------
// Attention block for MI455X (gfx1250, wave32, WMMA).
// Pipeline (all bf16 matrix math via v_wmma_f32_16x16x32_bf16):
//   1. f32 -> bf16 conversion of x / wq / wk / wv / wo
//   2. GEMM (A[M,K] @ B[N,K]^T) for Q, K, V projections (f32 accum out)
//   3. RoPE on Q,K (f32 -> bf16); V transposed to [d][s] bf16
//   4. Flash attention: 1 wave per (head, 16 query rows), key tiles of 32,
//      online softmax, P moved C-layout -> A-layout through LDS
//   5. GEMM attn @ wo^T -> f32 d_out
// ---------------------------------------------------------------------------

typedef __attribute__((ext_vector_type(16))) __bf16 bf16x16;
typedef __attribute__((ext_vector_type(8)))  __bf16 bf16x8;
typedef __attribute__((ext_vector_type(8)))  float  f32x8;

constexpr int kS   = 2048;
constexpr int kDim = 4096;
constexpr int kH   = 32;
constexpr int kKVH = 8;
constexpr int kD   = 128;

// ---- fragment loaders (layouts from CDNA5 ISA §7.12.2) ----------
// A (16x32 bf16): lane m=l%16, half h=l/16; elems 0..7 -> K=8h+0..7,
//                 elems 8..15 -> K=16+8h+0..7
__device__ __forceinline__ bf16x16 load_frag_a(const __bf16* base, int ld) {
    int lane = threadIdx.x & 31;
    int m = lane & 15, h = lane >> 4;
    const __bf16* p = base + (size_t)m * ld + h * 8;
    union { bf16x16 v; bf16x8 q[2]; } u;
    u.q[0] = *(const bf16x8*)(p);
    u.q[1] = *(const bf16x8*)(p + 16);
    return u.v;
}
// B (32x16 bf16): lane n=l%16, half h=l/16; elems 0..7 -> K=8h+0..7,
//                 elems 8..15 -> K=16+8h+0..7
__device__ __forceinline__ bf16x16 load_frag_b(const __bf16* base, int ld) {
    int lane = threadIdx.x & 31;
    int n = lane & 15, h = lane >> 4;
    const __bf16* p = base + (size_t)n * ld + h * 8;
    union { bf16x16 v; bf16x8 q[2]; } u;
    u.q[0] = *(const bf16x8*)(p);
    u.q[1] = *(const bf16x8*)(p + 16);
    return u.v;
}

// ---- elementwise converts ---------------------------------------------------
__global__ void cvt_f32_bf16(const float* __restrict__ in,
                             __bf16* __restrict__ out, size_t n) {
    for (size_t i = (size_t)blockIdx.x * blockDim.x + threadIdx.x; i < n;
         i += (size_t)gridDim.x * blockDim.x)
        out[i] = (__bf16)in[i];
}

// V [S][KVH*D] f32 -> Vt [KVH*D][S] bf16 (so PV B-fragments load contiguously)
__global__ void transpose_v(const float* __restrict__ v,
                            __bf16* __restrict__ vt) {
    size_t n = (size_t)kS * kKVH * kD;
    for (size_t i = (size_t)blockIdx.x * blockDim.x + threadIdx.x; i < n;
         i += (size_t)gridDim.x * blockDim.x) {
        size_t s = i / (kKVH * kD);
        size_t c = i % (kKVH * kD);
        vt[c * kS + s] = (__bf16)v[i];
    }
}

// RoPE: T [S][nh*D] f32 -> bf16, freqs_cis [S][D/2][2]
__global__ void rope_bf16(const float* __restrict__ t,
                          const float* __restrict__ fc,
                          __bf16* __restrict__ out, int nh) {
    size_t n = (size_t)kS * nh * (kD / 2);
    for (size_t i = (size_t)blockIdx.x * blockDim.x + threadIdx.x; i < n;
         i += (size_t)gridDim.x * blockDim.x) {
        int d2 = (int)(i % (kD / 2));
        size_t sh = i / (kD / 2);
        int hh = (int)(sh % nh);
        int s  = (int)(sh / nh);
        float c  = fc[(size_t)s * kD + 2 * d2 + 0];
        float si = fc[(size_t)s * kD + 2 * d2 + 1];
        size_t base = ((size_t)s * nh + hh) * kD + 2 * d2;
        float a = t[base], b = t[base + 1];
        out[base]     = (__bf16)(a * c - b * si);
        out[base + 1] = (__bf16)(a * si + b * c);
    }
}

// ---- GEMM: C[M,N] f32 = A[M,K] bf16 @ B[N,K]^T bf16 ------------------------
// 128 threads = 4 waves; block tile 128x128; wave tile 64x64 (4x4 WMMA frags)
__global__ __launch_bounds__(128, 1)
void gemm_bf16_nt(const __bf16* __restrict__ A, const __bf16* __restrict__ B,
                  float* __restrict__ C, int M, int N, int K) {
    int wave = threadIdx.x >> 5;
    int lane = threadIdx.x & 31;
    int wm = (wave >> 1) * 64 + blockIdx.y * 128;
    int wn = (wave & 1)  * 64 + blockIdx.x * 128;
    f32x8 acc[4][4] = {};
    for (int k0 = 0; k0 < K; k0 += 32) {
        bf16x16 af[4], bfr[4];
        #pragma unroll
        for (int i = 0; i < 4; ++i)
            af[i] = load_frag_a(A + (size_t)(wm + i * 16) * K + k0, K);
        #pragma unroll
        for (int j = 0; j < 4; ++j)
            bfr[j] = load_frag_b(B + (size_t)(wn + j * 16) * K + k0, K);
        #pragma unroll
        for (int i = 0; i < 4; ++i)
            #pragma unroll
            for (int j = 0; j < 4; ++j)
                acc[i][j] = __builtin_amdgcn_wmma_f32_16x16x32_bf16(
                    false, af[i], false, bfr[j], (short)0, acc[i][j],
                    false, false);
    }
    int n = lane & 15, h = lane >> 4;
    #pragma unroll
    for (int i = 0; i < 4; ++i)
        #pragma unroll
        for (int j = 0; j < 4; ++j)
            #pragma unroll
            for (int r = 0; r < 8; ++r) {
                size_t row = (size_t)(wm + i * 16 + r + 8 * h);
                size_t col = (size_t)(wn + j * 16 + n);
                C[row * N + col] = acc[i][j][r];
            }
}

// ---- Flash attention --------------------------------------------------------
// grid (S/16, H), block = 32 (one wave). Q rows qbase..qbase+15 for one head.
// Key tiles of 32 (two 16x16 score WMMA tiles per step, K-dim = D = 128).
__global__ __launch_bounds__(32, 1)
void flash_attn(const __bf16* __restrict__ Q,   // [S][H*D]
                const __bf16* __restrict__ Kc,  // [S][KVH*D]
                const __bf16* __restrict__ Vt,  // [KVH*D][S]
                __bf16* __restrict__ Oo) {      // [S][H*D]
    __shared__ __align__(16) __bf16 pbuf[16 * 32];
    const int lane = threadIdx.x & 31;
    const int n = lane & 15, h = lane >> 4;
    const int qbase = blockIdx.x * 16;
    const int head  = blockIdx.y;
    const int kvh   = head / (kH / kKVH);
    const float sscale = 11.313708499f;  // reference: scores / D**-0.5

    // persistent Q A-fragments over D=128 (4 k-steps of 32)
    bf16x16 qf[4];
    const __bf16* qrow = Q + (size_t)qbase * (kH * kD) + head * kD;
    #pragma unroll
    for (int kk = 0; kk < 4; ++kk)
        qf[kk] = load_frag_a(qrow + kk * 32, kH * kD);

    f32x8 acc[8] = {};               // O: 8 column blocks of 16 over D=128
    float mrun[8], lrun[8];
    #pragma unroll
    for (int r = 0; r < 8; ++r) { mrun[r] = -1e30f; lrun[r] = 0.f; }

    const int nkt = (qbase + 15) / 32 + 1;  // causal: kbase <= qbase+15
    for (int kt = 0; kt < nkt; ++kt) {
        const int kbase = kt * 32;
        f32x8 sc[2] = {};
        #pragma unroll
        for (int t = 0; t < 2; ++t) {
            const __bf16* krow =
                Kc + (size_t)(kbase + t * 16) * (kKVH * kD) + kvh * kD;
            #pragma unroll
            for (int kk = 0; kk < 4; ++kk) {
                bf16x16 kf = load_frag_b(krow + kk * 32, kKVH * kD);
                sc[t] = __builtin_amdgcn_wmma_f32_16x16x32_bf16(
                    false, qf[kk], false, kf, (short)0, sc[t], false, false);
            }
        }
        // scale + causal mask + row max (row M=r+8h lives in 16 lanes of a half)
        float rmax[8];
        #pragma unroll
        for (int r = 0; r < 8; ++r) {
            int q = qbase + r + 8 * h;
            float a0 = sc[0][r] * sscale;
            float a1 = sc[1][r] * sscale;
            if (kbase + n > q)      a0 = -1e30f;
            if (kbase + 16 + n > q) a1 = -1e30f;
            sc[0][r] = a0; sc[1][r] = a1;
            float mx = fmaxf(a0, a1);
            #pragma unroll
            for (int msk = 8; msk >= 1; msk >>= 1)
                mx = fmaxf(mx, __shfl_xor(mx, msk, 32));
            rmax[r] = mx;
        }
        // online softmax update
        float alpha[8];
        #pragma unroll
        for (int r = 0; r < 8; ++r) {
            float mnew = fmaxf(mrun[r], rmax[r]);
            alpha[r] = __expf(mrun[r] - mnew);
            mrun[r] = mnew;
            float p0 = __expf(sc[0][r] - mnew);
            float p1 = __expf(sc[1][r] - mnew);
            sc[0][r] = p0; sc[1][r] = p1;
            float rs = p0 + p1;
            #pragma unroll
            for (int msk = 8; msk >= 1; msk >>= 1)
                rs += __shfl_xor(rs, msk, 32);
            lrun[r] = lrun[r] * alpha[r] + rs;
        }
        #pragma unroll
        for (int cb = 0; cb < 8; ++cb)
            #pragma unroll
            for (int r = 0; r < 8; ++r)
                acc[cb][r] *= alpha[r];
        // P: C-layout -> A-layout via LDS (16x32 bf16 row-major)
        #pragma unroll
        for (int t = 0; t < 2; ++t)
            #pragma unroll
            for (int r = 0; r < 8; ++r)
                pbuf[(r + 8 * h) * 32 + t * 16 + n] = (__bf16)sc[t][r];
        union { bf16x16 v; bf16x8 q[2]; } pa;
        pa.q[0] = *(const bf16x8*)&pbuf[(lane & 15) * 32 + h * 8];
        pa.q[1] = *(const bf16x8*)&pbuf[(lane & 15) * 32 + 16 + h * 8];
        // O += P @ V  (B-frag: column d = cb*16+n, contiguous k in Vt)
        #pragma unroll
        for (int cb = 0; cb < 8; ++cb) {
            const __bf16* vcol =
                Vt + ((size_t)kvh * kD + cb * 16) * kS + kbase;
            bf16x16 vf = load_frag_b(vcol, kS);
            acc[cb] = __builtin_amdgcn_wmma_f32_16x16x32_bf16(
                false, pa.v, false, vf, (short)0, acc[cb], false, false);
        }
    }
    // normalize + store bf16 attn output [S][H*D]
    #pragma unroll
    for (int cb = 0; cb < 8; ++cb)
        #pragma unroll
        for (int r = 0; r < 8; ++r) {
            size_t row = (size_t)(qbase + r + 8 * h);
            size_t col = (size_t)head * kD + cb * 16 + n;
            Oo[row * (kH * kD) + col] = (__bf16)(acc[cb][r] / lrun[r]);
        }
}

// ---------------------------------------------------------------------------
extern "C" void kernel_launch(void* const* d_in, const int* in_sizes, int n_in,
                              void* d_out, int out_size, void* d_ws,
                              size_t ws_size, hipStream_t stream) {
    const float* x  = (const float*)d_in[0];
    const float* wq = (const float*)d_in[1];
    const float* wk = (const float*)d_in[2];
    const float* wv = (const float*)d_in[3];
    const float* wo = (const float*)d_in[4];
    const float* fc = (const float*)d_in[5];
    float* out = (float*)d_out;

    char* p = (char*)d_ws;
    auto take = [&](size_t bytes) {
        char* r = p;
        p += (bytes + 255) & ~(size_t)255;
        return r;
    };
    const size_t nX  = (size_t)kS * kDim;            // 8M
    const size_t nWQ = (size_t)kH * kD * kDim;       // 16M
    const size_t nWK = (size_t)kKVH * kD * kDim;     // 4M
    __bf16* xb  = (__bf16*)take(nX * 2);
    __bf16* wqb = (__bf16*)take(nWQ * 2);
    __bf16* wkb = (__bf16*)take(nWK * 2);
    __bf16* wvb = (__bf16*)take(nWK * 2);
    __bf16* wob = (__bf16*)take(nWQ * 2);
    float*  qf  = (float*)take((size_t)kS * kH * kD * 4);
    float*  kf  = (float*)take((size_t)kS * kKVH * kD * 4);
    float*  vf  = (float*)take((size_t)kS * kKVH * kD * 4);
    __bf16* qb  = (__bf16*)take((size_t)kS * kH * kD * 2);
    __bf16* kb  = (__bf16*)take((size_t)kS * kKVH * kD * 2);
    __bf16* vtb = (__bf16*)take((size_t)kS * kKVH * kD * 2);
    __bf16* ab  = (__bf16*)take((size_t)kS * kH * kD * 2);

    // 1. convert inputs to bf16
    cvt_f32_bf16<<<2048, 256, 0, stream>>>(x,  xb,  nX);
    cvt_f32_bf16<<<2048, 256, 0, stream>>>(wq, wqb, nWQ);
    cvt_f32_bf16<<<2048, 256, 0, stream>>>(wk, wkb, nWK);
    cvt_f32_bf16<<<2048, 256, 0, stream>>>(wv, wvb, nWK);
    cvt_f32_bf16<<<2048, 256, 0, stream>>>(wo, wob, nWQ);

    // 2. projections (C = x @ W^T)
    dim3 blk(128);
    gemm_bf16_nt<<<dim3(kH * kD / 128, kS / 128), blk, 0, stream>>>(
        xb, wqb, qf, kS, kH * kD, kDim);
    gemm_bf16_nt<<<dim3(kKVH * kD / 128, kS / 128), blk, 0, stream>>>(
        xb, wkb, kf, kS, kKVH * kD, kDim);
    gemm_bf16_nt<<<dim3(kKVH * kD / 128, kS / 128), blk, 0, stream>>>(
        xb, wvb, vf, kS, kKVH * kD, kDim);

    // 3. RoPE (Q, K) and V transpose, all -> bf16
    rope_bf16<<<1024, 256, 0, stream>>>(qf, fc, qb, kH);
    rope_bf16<<<1024, 256, 0, stream>>>(kf, fc, kb, kKVH);
    transpose_v<<<1024, 256, 0, stream>>>(vf, vtb);

    // 4. flash attention: one wave per (16 query rows, head)
    flash_attn<<<dim3(kS / 16, kH), dim3(32), 0, stream>>>(qb, kb, vtb, ab);

    // 5. output projection -> f32 d_out
    gemm_bf16_nt<<<dim3(kDim / 128, kS / 128), blk, 0, stream>>>(
        ab, wob, out, kS, kDim, kH * kD);
}
